// SelfAttention_33801392619888
// MI455X (gfx1250) — matrix operand order repaired and checked by reading, not compiled, not run on hardware
//
#include <hip/hip_runtime.h>
#include <hip/hip_bf16.h>
#include <stdint.h>

// Sizes (fixed by the reference)
#define BB   32
#define NN   1024
#define CC   1280
#define CQK  160
#define OUTD 1024

#define KTILE 32
#define LDSK  40   // LDS row stride in halves (padded; 80B => 16B-aligned rows)

typedef __attribute__((ext_vector_type(16))) _Float16 v16h;
typedef __attribute__((ext_vector_type(8)))  float    v8f;

union HFrag { v16h v; unsigned u[8]; };

// ---- CDNA5 async global->LDS copy (ASYNCcnt-tracked DMA) -------------------
// LDS byte address = low 32 bits of the generic shared-aperture address
// (ISA 10.2).
__device__ __forceinline__ void async_b128(const _Float16* g, _Float16* l) {
    unsigned loff = (unsigned)(uintptr_t)l;
    asm volatile("global_load_async_to_lds_b128 %0, %1, off"
                 :: "v"(loff), "v"(g) : "memory");
}
__device__ __forceinline__ void wait_async0() {
    asm volatile("s_wait_asynccnt 0x0" ::: "memory");
}

// A fragment (16x32 f16, MxK), ISA 7.12.2 wave32 layout.
__device__ __forceinline__ v16h load_frag_A(const _Float16* rowPtr, int lane) {
    HFrag f;
    const int half = lane >> 4;
    const unsigned* p = (const unsigned*)rowPtr;  // 2 halves per 32b
#pragma unroll
    for (int j = 0; j < 8; ++j) {
        int kk = ((j < 4) ? (2 * j) : (16 + 2 * (j - 4))) + half * 8;
        f.u[j] = p[kk >> 1];
    }
    return f.v;
}

// B fragment (32x16 f16, KxN): lane%16 = N; K order matches A (0-7,16-23 | 8-15,24-31).
__device__ __forceinline__ v16h load_frag_B(const _Float16* colPtr, int lane) {
    HFrag f;
    const int half = lane >> 4;
    const unsigned* p = (const unsigned*)colPtr;
#pragma unroll
    for (int j = 0; j < 8; ++j) {
        int kk = ((j < 4) ? (2 * j) : (16 + 2 * (j - 4))) + half * 8;
        f.u[j] = p[kk >> 1];
    }
    return f.v;
}

// ---- generic f16 WMMA GEMM: D = A(MxK) . B(NcxK)^T  (+bias | g*acc+resid) --
// Block = 256 threads = 8 waves arranged WR x WC; each wave computes RT x CT
// 16x16 WMMA tiles. Block tile = (WR*RT*16) x (WC*CT*16). Double-buffered LDS
// with async global->LDS staging fully overlapped with WMMA compute; last
// K-iteration peeled so the steady-state loop is branch-free.
// TRANSD: store D transposed ([Nc][M], ldd = M-stride), used to produce vT.
template<int WR, int WC, int RT, int CT, typename TOUT, bool TRANSD>
__global__ __launch_bounds__(256) void gemm_wmma(
    const _Float16* __restrict__ A,  long long sAb, int lda,
    const _Float16* __restrict__ Bm, long long sBb, int ldb,
    const float* __restrict__ bias,
    const float* __restrict__ resid, long long sRb, int ldr,
    const float* __restrict__ gammap,
    TOUT* __restrict__ D, long long sDb, int ldd,
    int K)
{
    constexpr int MT  = WR * RT * 16;
    constexpr int NT  = WC * CT * 16;
    constexpr int ACH = (MT * 4 + 255) / 256;  // 16B async chunks / thread (A)
    constexpr int BCH = (NT * 4 + 255) / 256;  // 16B async chunks / thread (B)
    __shared__ _Float16 sA[2][MT * LDSK];
    __shared__ _Float16 sB[2][NT * LDSK];

    const int b    = blockIdx.z;
    const int row0 = blockIdx.y * MT;
    const int col0 = blockIdx.x * NT;

    const _Float16* Ab = A  + (size_t)b * sAb + (size_t)row0 * lda;
    const _Float16* Bb = Bm + (size_t)b * sBb + (size_t)col0 * ldb;
    TOUT*           Db = D  + (size_t)b * sDb;

    const int t    = threadIdx.x;
    const int wid  = t >> 5;
    const int lane = t & 31;
    const int wr   = wid % WR;
    const int wc   = wid / WR;

    v8f acc[RT][CT] = {};

    // ---- hoisted per-thread staging addresses (advance by KTILE per stage) -
    const _Float16* aG[ACH]; int aL[ACH];
#pragma unroll
    for (int i = 0; i < ACH; ++i) {
        int c = (t + i * 256) % (MT * 4);
        int r = c >> 2, kk = (c & 3) * 8;
        aG[i] = Ab + (size_t)r * lda + kk;
        aL[i] = r * LDSK + kk;
    }
    const _Float16* bG[BCH]; int bL[BCH];
#pragma unroll
    for (int i = 0; i < BCH; ++i) {
        int c = (t + i * 256) % (NT * 4);   // wraps for NT=32 (benign dup copy)
        int r = c >> 2, kk = (c & 3) * 8;
        bG[i] = Bb + (size_t)r * ldb + kk;
        bL[i] = r * LDSK + kk;
    }

    auto stage = [&](int buf) {
#pragma unroll
        for (int i = 0; i < ACH; ++i) {
            async_b128(aG[i], &sA[buf][aL[i]]);
            aG[i] += KTILE;
        }
#pragma unroll
        for (int i = 0; i < BCH; ++i) {
            async_b128(bG[i], &sB[buf][bL[i]]);
            bG[i] += KTILE;
        }
    };

    const int m0 = wr * RT * 16 + (lane & 15);
    const int n0 = wc * CT * 16 + (lane & 15);

    auto compute = [&](int cur) {
        const _Float16* bufA = sA[cur];
        const _Float16* bufB = sB[cur];
        v16h af[RT], bf[CT];
#pragma unroll
        for (int rt = 0; rt < RT; ++rt)
            af[rt] = load_frag_A(bufA + (m0 + rt * 16) * LDSK, lane);
#pragma unroll
        for (int ct = 0; ct < CT; ++ct)
            bf[ct] = load_frag_B(bufB + (n0 + ct * 16) * LDSK, lane);
#pragma unroll
        for (int rt = 0; rt < RT; ++rt)
#pragma unroll
            for (int ct = 0; ct < CT; ++ct)
                acc[rt][ct] = __builtin_amdgcn_wmma_f32_16x16x32_f16(
                    false, af[rt], false, bf[ct], (short)0, acc[rt][ct],
                    false, false);
    };

    const int KT = K / KTILE;

    stage(0);                 // DMA tile 0
    wait_async0();
    __syncthreads();

    for (int kt = 0; kt < KT - 1; ++kt) {   // branch-free steady state
        const int cur = kt & 1;
        stage(cur ^ 1);       // DMA tile kt+1 (overlaps with WMMAs below)
        compute(cur);
        wait_async0();
        __syncthreads();
    }
    compute((KT - 1) & 1);    // peeled tail: no staging, no barrier needed

    // Epilogue. C/D layout: VGPR j -> M = j (+8 for lanes 16..31), N = lane%16.
    const int half = lane >> 4;
    const int n    = lane & 15;
    const float g  = gammap ? gammap[0] : 1.0f;

#pragma unroll
    for (int rt = 0; rt < RT; ++rt) {
        const int grb = row0 + wr * RT * 16 + rt * 16 + half * 8;
#pragma unroll
        for (int ct = 0; ct < CT; ++ct) {
            const int gc   = col0 + wc * CT * 16 + ct * 16 + n;
            const float bv = bias ? bias[gc] : 0.0f;
            const float* rp =
                resid ? (resid + (size_t)b * sRb + gc) : nullptr;
#pragma unroll
            for (int j = 0; j < 8; ++j) {
                const int gr = grb + j;
                float v = acc[rt][ct][j];
                v += bv;
                if (resid) v = g * acc[rt][ct][j] + rp[(size_t)gr * ldr];
                if (TRANSD)
                    Db[(size_t)gc * ldd + gr] = (TOUT)v;   // contiguous per lane
                else
                    Db[(size_t)gr * ldd + gc] = (TOUT)v;
            }
        }
    }
}

// ---- f32 -> f16 pack -------------------------------------------------------
__global__ __launch_bounds__(256) void pack_f16(
    const float* __restrict__ src, _Float16* __restrict__ dst, long long n)
{
    long long i = ((long long)blockIdx.x * 256 + threadIdx.x) * 4;
    if (i + 3 < n) {
        float4 v = *(const float4*)(src + i);
        dst[i + 0] = (_Float16)v.x;
        dst[i + 1] = (_Float16)v.y;
        dst[i + 2] = (_Float16)v.z;
        dst[i + 3] = (_Float16)v.w;
    }
}

// ---- masked softmax (f16 in/out, f32 math), last dim = 1024, in place ------
__global__ __launch_bounds__(256) void softmax_mask_kernel(
    _Float16* __restrict__ scores, const float* __restrict__ mask)
{
    const int row = blockIdx.x;
    const int b   = blockIdx.y;
    _Float16* s = scores + ((size_t)b * NN + row) * NN;
    const float* mk = mask + (size_t)b * 64 * 64;

    const int t    = threadIdx.x;
    const int lane = t & 31;
    const int wid  = t >> 5;
    __shared__ float red[8];

    float v[4];
    float mx = -3.0e38f;
#pragma unroll
    for (int i = 0; i < 4; ++i) {
        int col  = t + 256 * i;
        float mv = mk[((col >> 5) * 2) * 64 + (col & 31) * 2];
        float x  = (float)s[col] - (1.0f - mv) * 1.0e6f;
        v[i] = x;
        mx   = fmaxf(mx, x);
    }
#pragma unroll
    for (int o = 16; o >= 1; o >>= 1) mx = fmaxf(mx, __shfl_xor(mx, o, 32));
    if (lane == 0) red[wid] = mx;
    __syncthreads();
    float m2 = red[0];
#pragma unroll
    for (int i = 1; i < 8; ++i) m2 = fmaxf(m2, red[i]);

    float sum = 0.0f;
#pragma unroll
    for (int i = 0; i < 4; ++i) { v[i] = __expf(v[i] - m2); sum += v[i]; }
#pragma unroll
    for (int o = 16; o >= 1; o >>= 1) sum += __shfl_xor(sum, o, 32);
    __syncthreads();
    if (lane == 0) red[wid] = sum;
    __syncthreads();
    float tot = 0.0f;
#pragma unroll
    for (int i = 0; i < 8; ++i) tot += red[i];
    float inv = 1.0f / tot;
#pragma unroll
    for (int i = 0; i < 4; ++i) s[t + 256 * i] = (_Float16)(v[i] * inv);
}

extern "C" void kernel_launch(void* const* d_in, const int* in_sizes, int n_in,
                              void* d_out, int out_size, void* d_ws, size_t ws_size,
                              hipStream_t stream) {
    const float* x     = (const float*)d_in[0];
    const float* mask  = (const float*)d_in[1];
    const float* Wq    = (const float*)d_in[2];
    const float* bq    = (const float*)d_in[3];
    const float* Wk    = (const float*)d_in[4];
    const float* bk    = (const float*)d_in[5];
    const float* Wv    = (const float*)d_in[6];
    const float* bv    = (const float*)d_in[7];
    const float* gamma = (const float*)d_in[8];
    const float* Wp    = (const float*)d_in[9];
    const float* bp    = (const float*)d_in[10];
    float* out = (float*)d_out;

    // f16 workspace layout (~346 MB)
    _Float16* h = (_Float16*)d_ws;
    const long long X_SZ  = (long long)BB * NN * CC;   // 41,943,040
    const long long WQ_SZ = (long long)CQK * CC;       //    204,800
    const long long WV_SZ = (long long)CC * CC;        //  1,638,400
    const long long WP_SZ = (long long)OUTD * CC;      //  1,310,720
    const long long QK_SZ = (long long)BB * NN * CQK;  //  5,242,880
    const long long S_SZ  = (long long)BB * NN * NN;   // 33,554,432
    _Float16* x16  = h;             h += X_SZ;
    _Float16* wq16 = h;             h += WQ_SZ;
    _Float16* wk16 = h;             h += WQ_SZ;
    _Float16* wv16 = h;             h += WV_SZ;
    _Float16* wp16 = h;             h += WP_SZ;
    _Float16* q16  = h;             h += QK_SZ;
    _Float16* k16  = h;             h += QK_SZ;
    _Float16* vT16 = h;             h += X_SZ;         // [B][C][N] transposed
    _Float16* s16  = h;             h += S_SZ;
    _Float16* o16  = h;             /* h += X_SZ; */

    const dim3 blk(256);
    const long long sX = (long long)NN * CC;   // x / vT / out2 batch stride
    const long long sQ = (long long)NN * CQK;  // q / k batch stride
    const long long sS = (long long)NN * NN;   // scores batch stride

    // 0) pack inputs to f16
    pack_f16<<<dim3((unsigned)(X_SZ  / 1024)), blk, 0, stream>>>(x,  x16,  X_SZ);
    pack_f16<<<dim3((unsigned)(WQ_SZ / 1024)), blk, 0, stream>>>(Wq, wq16, WQ_SZ);
    pack_f16<<<dim3((unsigned)(WQ_SZ / 1024)), blk, 0, stream>>>(Wk, wk16, WQ_SZ);
    pack_f16<<<dim3((unsigned)(WV_SZ / 1024)), blk, 0, stream>>>(Wv, wv16, WV_SZ);
    pack_f16<<<dim3((unsigned)(WP_SZ / 1024)), blk, 0, stream>>>(Wp, wp16, WP_SZ);

    // 1) q = x @ Wq^T + bq   [B,1024,160]   (small-N config: 128x32 tiles)
    gemm_wmma<8, 1, 1, 2, _Float16, false>
        <<<dim3(CQK / 32, NN / 128, BB), blk, 0, stream>>>(
        x16, sX, CC, wq16, 0, CC, bq, nullptr, 0, 0, nullptr,
        q16, sQ, CQK, CC);
    // 2) k = x @ Wk^T + bk
    gemm_wmma<8, 1, 1, 2, _Float16, false>
        <<<dim3(CQK / 32, NN / 128, BB), blk, 0, stream>>>(
        x16, sX, CC, wk16, 0, CC, bk, nullptr, 0, 0, nullptr,
        k16, sQ, CQK, CC);
    // 3) vT = (x @ Wv^T + bv)^T  stored [B][c][m]  (128x128 tiles, TRANSD)
    gemm_wmma<4, 2, 2, 4, _Float16, true>
        <<<dim3(CC / 128, NN / 128, BB), blk, 0, stream>>>(
        x16, sX, CC, wv16, 0, CC, bv, nullptr, 0, 0, nullptr,
        vT16, sX, NN, CC);
    // 4) scores = q @ k^T   [B,1024,1024], K=160
    gemm_wmma<4, 2, 2, 4, _Float16, false>
        <<<dim3(NN / 128, NN / 128, BB), blk, 0, stream>>>(
        q16, sQ, CQK, k16, sQ, CQK, nullptr, nullptr, 0, 0, nullptr,
        s16, sS, NN, CQK);
    // 5) masked softmax in place
    softmax_mask_kernel<<<dim3(NN, BB), blk, 0, stream>>>(s16, mask);
    // 6) out2 = gamma * (weights @ v) + x ; B operand = vT (k=m contiguous)
    gemm_wmma<4, 2, 2, 4, _Float16, false>
        <<<dim3(CC / 128, NN / 128, BB), blk, 0, stream>>>(
        s16, sS, NN, vT16, sX, NN, nullptr, x, sX, CC, gamma,
        o16, sX, CC, NN);
    // 7) out = out2 @ Wp^T + bp   [B,1024,1024] f32
    gemm_wmma<4, 2, 2, 4, float, false>
        <<<dim3(OUTD / 128, NN / 128, BB), blk, 0, stream>>>(
        o16, sX, CC, wp16, 0, CC, bp, nullptr, 0, 0, nullptr,
        out, (long long)NN * OUTD, OUTD, CC);
}
